// AttentionMIL_33775622815978
// MI455X (gfx1250) — matrix operand order repaired and checked by reading, not compiled, not run on hardware
//
#include <hip/hip_runtime.h>
#include <math.h>

// Problem constants (reference: B,N,F,L = 16,16384,256,128)
#define B_   16
#define N_   16384
#define F_   256
#define L_   128
#define ROWS 128                 // rows (instances) per workgroup
#define WGPB (N_ / ROWS)         // 128 workgroups per batch
#define KT   (F_ / 32)           // 8 k-tiles of 32 (bf16 WMMA K)
#define LT   (L_ / 16)           // 8 l-tiles of 16
#define FRAG_PER_MAT (KT * LT * 32)   // 2048 v16bf entries per matrix

typedef __bf16 v16bf __attribute__((ext_vector_type(16)));
typedef float  v8f   __attribute__((ext_vector_type(8)));
typedef float  v4f   __attribute__((ext_vector_type(4)));

#if __has_builtin(__builtin_amdgcn_tanhf)
#define TANH_(x) __builtin_amdgcn_tanhf(x)
#else
#define TANH_(x) tanhf(x)
#endif

__device__ __forceinline__ float sigmoidf_(float x) {
    return 1.0f / (1.0f + __expf(-x));
}

// ---------------------------------------------------------------------------
// Kernel 0: one-time pre-swizzle of V and U into bf16 WMMA B-fragment layout.
// Entry (mat, kt, lt, ln): col n = lt*16 + (ln&15); elem j = K (kt*32 + (ln>>4)*16 + j)
// ---------------------------------------------------------------------------
__global__ __launch_bounds__(256)
void mil_swizzle(const float* __restrict__ V,
                 const float* __restrict__ U,
                 v16bf* __restrict__ gFrag)     // [2][KT][LT][32]
{
    const int id = blockIdx.x * 256 + threadIdx.x;   // [0, 2*FRAG_PER_MAT)
    const int mat = id >> 11;                        // FRAG_PER_MAT = 2048
    const int rem = id & 2047;
    const int kt  = rem >> 8;
    const int r2  = rem & 255;
    const int lt  = r2 >> 5;
    const int ln  = r2 & 31;
    const int n   = lt * 16 + (ln & 15);
    const int kb  = kt * 32 + (ln >> 4) * 16;
    const float* src = (mat ? U : V) + (size_t)kb * L_ + n;
    v16bf frag;
    #pragma unroll
    for (int j = 0; j < 16; ++j)
        frag[j] = (__bf16)src[(size_t)j * L_];
    gFrag[id] = frag;
}

// ---------------------------------------------------------------------------
// Kernel 1: fused scores (two bf16 WMMA GEMMs + gate) + online-softmax partial
// pooling. One HBM pass over x. Double-buffered LDS staging (1 barrier / k-tile).
// ---------------------------------------------------------------------------
__global__ __launch_bounds__(256)
void mil_score_pool(const float* __restrict__ x,
                    const v16bf* __restrict__ gFrag,  // pre-swizzled V/U frags
                    const float* __restrict__ w,
                    float* __restrict__ scores,   // [B*N] raw scores
                    float* __restrict__ zPart,    // [B*WGPB, F] partial weighted sums
                    float* __restrict__ mPart,    // [B*WGPB] tile max
                    float* __restrict__ dPart)    // [B*WGPB] tile exp-sum
{
    __shared__ v16bf sV[2][LT][32];   // ping-pong staged B fragments (V)
    __shared__ v16bf sU[2][LT][32];   // ping-pong staged B fragments (U)
    __shared__ float sS[ROWS];        // per-row scores -> exp weights
    __shared__ float sRed[16];

    const int bid  = blockIdx.x;
    const int b    = bid / WGPB;
    const int wg   = bid % WGPB;
    const int rowBase = wg * ROWS;
    const int tid  = threadIdx.x;
    const int wave = tid >> 5;
    const int lane = tid & 31;
    const int m16  = lane & 15;
    const int half = lane >> 4;

    const float* xb = x + (size_t)b * N_ * F_;
    const v16bf* gV = gFrag;                  // [KT][256]
    const v16bf* gU = gFrag + FRAG_PER_MAT;   // [KT][256]

    // f32 accumulators: 8 l-tiles for each of the two GEMM branches
    v8f accV[LT], accU[LT];
    #pragma unroll
    for (int i = 0; i < LT; ++i) { accV[i] = {}; accU[i] = {}; }

    const size_t waveRow = (size_t)(rowBase + wave * 16 + m16);

    // stage k-tile 0 into buffer 0 (256 threads -> 1 V + 1 U entry each)
    (&sV[0][0][0])[tid] = gV[tid];
    (&sU[0][0][0])[tid] = gU[tid];
    __syncthreads();

    for (int kt = 0; kt < KT; ++kt) {
        const int buf = kt & 1;

        // ---- stage NEXT k-tile into the other buffer (overlaps with WMMAs) ----
        if (kt + 1 < KT) {
            (&sV[buf ^ 1][0][0])[tid] = gV[(kt + 1) * 256 + tid];
            (&sU[buf ^ 1][0][0])[tid] = gU[(kt + 1) * 256 + tid];
        }

        // ---- A fragment: 16 rows x 32 K of x, converted to bf16 ----
        // elems 0..15 = x[row][kb + half*16 + j]
        const float* ap = xb + waveRow * F_ + kt * 32 + half * 16;
        v4f a0 = *(const v4f*)(ap + 0);
        v4f a1 = *(const v4f*)(ap + 4);
        v4f a2 = *(const v4f*)(ap + 8);
        v4f a3 = *(const v4f*)(ap + 12);
        if (kt + 1 < KT)  // prefetch next k-tile slab of this wave's rows
            __builtin_prefetch(ap + 32, 0, 1);
        v16bf afrag;
        #pragma unroll
        for (int j = 0; j < 4; ++j) {
            afrag[j]      = (__bf16)a0[j];
            afrag[4 + j]  = (__bf16)a1[j];
            afrag[8 + j]  = (__bf16)a2[j];
            afrag[12 + j] = (__bf16)a3[j];
        }

        // ---- 16 WMMAs, B fragments register-double-buffered so ds_loads for
        //      l-tile lt+1 are in flight while WMMAs of lt execute ----
        v16bf bv = sV[buf][0][lane];
        v16bf bu = sU[buf][0][lane];
        #pragma unroll
        for (int lt = 0; lt < LT; ++lt) {
            v16bf bvn, bun;
            if (lt + 1 < LT) {
                bvn = sV[buf][lt + 1][lane];
                bun = sU[buf][lt + 1][lane];
            }
            accV[lt] = __builtin_amdgcn_wmma_f32_16x16x32_bf16(
                false, afrag, false, bv, (short)0, accV[lt], false, false);
            accU[lt] = __builtin_amdgcn_wmma_f32_16x16x32_bf16(
                false, afrag, false, bu, (short)0, accU[lt], false, false);
            if (lt + 1 < LT) { bv = bvn; bu = bun; }
        }
        __syncthreads();   // staging of kt+1 done; reads of buf done before kt+2 overwrites it
    }

    // ---- gate + dot with w: s[m] = sum_l tanh(xV)[m,l]*sigmoid(xU)[m,l]*w[l] ----
    // C/D layout: lane holds col l = lt*16 + (lane&15); VGPR v holds row v + 8*half.
    float part[8];
    #pragma unroll
    for (int v = 0; v < 8; ++v) part[v] = 0.0f;
    #pragma unroll
    for (int lt = 0; lt < LT; ++lt) {
        const float wv = w[lt * 16 + m16];
        #pragma unroll
        for (int v = 0; v < 8; ++v) {
            float g = TANH_(accV[lt][v]) * sigmoidf_(accU[lt][v]);
            part[v] += g * wv;
        }
    }
    // reduce over the 16 lanes of each half (l-dimension)
    #pragma unroll
    for (int v = 0; v < 8; ++v) {
        float p = part[v];
        p += __shfl_xor(p, 1, 32);
        p += __shfl_xor(p, 2, 32);
        p += __shfl_xor(p, 4, 32);
        p += __shfl_xor(p, 8, 32);
        part[v] = p;
    }
    if (m16 == 0) {  // one lane per half writes its 8 rows
        #pragma unroll
        for (int v = 0; v < 8; ++v) {
            const int r = wave * 16 + 8 * half + v;
            sS[r] = part[v];
            scores[(size_t)b * N_ + rowBase + r] = part[v];
        }
    }
    __syncthreads();

    // ---- workgroup softmax partials: m = max, d = sum exp(s-m) ----
    float lm = (tid < ROWS) ? sS[tid] : -3.0e38f;
    #pragma unroll
    for (int msk = 16; msk >= 1; msk >>= 1)
        lm = fmaxf(lm, __shfl_xor(lm, msk, 32));
    if (lane == 0) sRed[wave] = lm;
    __syncthreads();
    if (tid == 0) {
        float mm = sRed[0];
        #pragma unroll
        for (int i = 1; i < 8; ++i) mm = fmaxf(mm, sRed[i]);
        sRed[0] = mm;
    }
    __syncthreads();
    const float mWG = sRed[0];

    float e = 0.0f;
    if (tid < ROWS) e = __expf(sS[tid] - mWG);
    float ls = e;
    #pragma unroll
    for (int msk = 16; msk >= 1; msk >>= 1)
        ls += __shfl_xor(ls, msk, 32);
    if (lane == 0) sRed[8 + wave] = ls;
    if (tid < ROWS) sS[tid] = e;   // sS now holds exp-weights
    __syncthreads();
    if (tid == 0) {
        float d = 0.0f;
        #pragma unroll
        for (int i = 0; i < 8; ++i) d += sRed[8 + i];
        mPart[bid] = mWG;
        dPart[bid] = d;
    }
    __syncthreads();

    // ---- partial pooling: z[f] = sum_r e[r] * x[r][f] (tile hits cache) ----
    float z = 0.0f;
    const float* xcol = xb + (size_t)rowBase * F_ + tid;
    for (int r = 0; r < ROWS; ++r)
        z += sS[r] * xcol[(size_t)r * F_];
    zPart[(size_t)bid * F_ + tid] = z;
}

// ---------------------------------------------------------------------------
// Kernel 2: merge per-WG partials with max-rescaling; write Z = (Σ c_i z_i)/D
// ---------------------------------------------------------------------------
__global__ __launch_bounds__(256)
void mil_combine(const float* __restrict__ zPart,
                 const float* __restrict__ mPart,
                 const float* __restrict__ dPart,
                 float* __restrict__ Zout,     // [B, F]
                 float* __restrict__ Mfin,     // [B]
                 float* __restrict__ Dfin)     // [B]
{
    const int b    = blockIdx.x;
    const int tid  = threadIdx.x;
    const int lane = tid & 31;
    const int wave = tid >> 5;
    __shared__ float sc[WGPB];
    __shared__ float red[16];

    float lm = (tid < WGPB) ? mPart[b * WGPB + tid] : -3.0e38f;
    if (tid < WGPB) sc[tid] = lm;
    #pragma unroll
    for (int msk = 16; msk >= 1; msk >>= 1)
        lm = fmaxf(lm, __shfl_xor(lm, msk, 32));
    if (lane == 0) red[wave] = lm;
    __syncthreads();
    if (tid == 0) {
        float mm = red[0];
        #pragma unroll
        for (int i = 1; i < 8; ++i) mm = fmaxf(mm, red[i]);
        red[0] = mm;
    }
    __syncthreads();
    const float M = red[0];

    float coef = 0.0f;
    if (tid < WGPB) coef = __expf(sc[tid] - M);
    float ld = (tid < WGPB) ? coef * dPart[b * WGPB + tid] : 0.0f;
    if (tid < WGPB) sc[tid] = coef;
    #pragma unroll
    for (int msk = 16; msk >= 1; msk >>= 1)
        ld += __shfl_xor(ld, msk, 32);
    if (lane == 0) red[8 + wave] = ld;
    __syncthreads();
    if (tid == 0) {
        float d = 0.0f;
        #pragma unroll
        for (int i = 0; i < 8; ++i) d += red[8 + i];
        red[8] = d;
    }
    __syncthreads();
    const float D = red[8];

    float z = 0.0f;
    for (int i = 0; i < WGPB; ++i)
        z += sc[i] * zPart[((size_t)(b * WGPB + i)) * F_ + tid];
    Zout[b * F_ + tid] = z / D;
    if (tid == 0) { Mfin[b] = M; Dfin[b] = D; }
}

// ---------------------------------------------------------------------------
// Kernel 3: A[b,n] = exp(s - M_b) / D_b
// ---------------------------------------------------------------------------
__global__ __launch_bounds__(256)
void mil_attn(const float* __restrict__ scores,
              const float* __restrict__ Mfin,
              const float* __restrict__ Dfin,
              float* __restrict__ Aout)       // [B*N]
{
    const int idx = blockIdx.x * 256 + threadIdx.x;  // B*N divisible by 256
    const int b   = idx >> 14;                       // N_ = 16384
    Aout[idx] = __expf(scores[idx] - Mfin[b]) / Dfin[b];
}

// ---------------------------------------------------------------------------
extern "C" void kernel_launch(void* const* d_in, const int* in_sizes, int n_in,
                              void* d_out, int out_size, void* d_ws, size_t ws_size,
                              hipStream_t stream) {
    (void)in_sizes; (void)n_in; (void)out_size; (void)ws_size;
    const float* x = (const float*)d_in[0];   // [B,N,F]
    const float* V = (const float*)d_in[1];   // [F,L]
    const float* U = (const float*)d_in[2];   // [F,L]
    const float* w = (const float*)d_in[3];   // [L,1]

    float* out  = (float*)d_out;
    float* Zout = out;                 // [B,F]  = 4096 floats
    float* Aout = out + B_ * F_;       // [B,N]  = 262144 floats

    float* ws     = (float*)d_ws;
    v16bf* gFrag  = (v16bf*)ws;                       // 2*2048 v16bf = 128 KB (32B aligned at ws start)
    float* fEnd   = ws + 2 * FRAG_PER_MAT * 8;        // 32768 floats
    float* scores = fEnd;                             // B*N          = 262144
    float* zPart  = scores + (size_t)B_ * N_;         // B*WGPB*F     = 524288
    float* mPart  = zPart + (size_t)B_ * WGPB * F_;   // B*WGPB       = 2048
    float* dPart  = mPart + (size_t)B_ * WGPB;        // B*WGPB       = 2048
    float* Mfin   = dPart + (size_t)B_ * WGPB;        // B            = 16
    float* Dfin   = Mfin + B_;                        // B            = 16

    mil_swizzle<<<(2 * FRAG_PER_MAT) / 256, 256, 0, stream>>>(V, U, gFrag);
    mil_score_pool<<<B_ * WGPB, 256, 0, stream>>>(x, gFrag, w,
                                                  scores, zPart, mPart, dPart);
    mil_combine<<<B_, 256, 0, stream>>>(zPart, mPart, dPart, Zout, Mfin, Dfin);
    mil_attn<<<(B_ * N_) / 256, 256, 0, stream>>>(scores, Mfin, Dfin, Aout);
}
